// QuantizedLinear_6133213299350
// MI455X (gfx1250) — compile-verified, matrix operand order checked by reading, not run on hardware
//
#include <hip/hip_runtime.h>

// QuantizedLinear: y[m,o] = scale[o] * (x[m,:] . w8[o,:]) + bias[o]
// M=8192 (B*S), N=4096 (DOUT), K=4096 (DIN).
// f16 WMMA path: int8 weights are exact in f16; scale applied post-accumulation.
// Double-buffered LDS: one barrier per K-tile; staging overlaps other waves' WMMAs.

typedef _Float16 v16h __attribute__((ext_vector_type(16)));
typedef _Float16 v8h  __attribute__((ext_vector_type(8)));
typedef float    v8f  __attribute__((ext_vector_type(8)));
typedef float    v4f  __attribute__((ext_vector_type(4)));
typedef int      v4i  __attribute__((ext_vector_type(4)));

namespace {
constexpr int Mtot = 8192;
constexpr int Ntot = 4096;
constexpr int Ktot = 4096;

constexpr int BM = 128, BN = 128, BK = 64;
constexpr int LDT = BK + 8;              // padded LDS row stride (f16 elems) -> 144B rows
constexpr int THREADS = 256;             // 8 wave32
constexpr int KT = Ktot / BK;            // 64 k-tiles
constexpr int GROUPS = (BM * BK) / 8;    // 1024 groups of 8 elements per tile
constexpr int GPT = GROUPS / THREADS;    // 4 groups per thread
}

__global__ __launch_bounds__(THREADS)
void qlinear_wmma_f16(const float* __restrict__ xg,
                      const int*   __restrict__ wg,
                      const float* __restrict__ scale,
                      const float* __restrict__ bias,
                      float*       __restrict__ out)
{
    // double-buffered tiles: 2 * (128*72 + 128*72) * 2B = 72 KB of 320 KB WGP LDS
    __shared__ __align__(16) _Float16 As[2][BM * LDT];
    __shared__ __align__(16) _Float16 Bs[2][BN * LDT];

    const int tid   = threadIdx.x;
    const int lane  = tid & 31;
    const int wave  = tid >> 5;
    const int waveM = wave & 3;          // 4 row-slices of 32
    const int waveN = wave >> 2;         // 2 col-slices of 64
    const int half  = lane >> 4;         // lane half selects K sub-offset
    const int lrow  = lane & 15;

    const int nBase = blockIdx.x * BN;
    const int mBase = blockIdx.y * BM;

    // register staging for the software pipeline
    float areg[GPT][8];
    int   breg[GPT][8];

    auto loadTiles = [&](int kt) {
#pragma unroll
        for (int g = 0; g < GPT; ++g) {
            const int G    = g * THREADS + tid;     // consecutive tid -> consecutive 32B
            const int row  = G >> 3;
            const int kofs = (G & 7) * 8;
            const float* ap = xg + (size_t)(mBase + row) * Ktot + kt * BK + kofs;
            const int*   bp = wg + (size_t)(nBase + row) * Ktot + kt * BK + kofs;
            v4f a0 = *(const v4f*)ap;
            v4f a1 = *(const v4f*)(ap + 4);
            v4i b0 = *(const v4i*)bp;
            v4i b1 = *(const v4i*)(bp + 4);
#pragma unroll
            for (int e = 0; e < 4; ++e) {
                areg[g][e] = a0[e]; areg[g][4 + e] = a1[e];
                breg[g][e] = b0[e]; breg[g][4 + e] = b1[e];
            }
        }
    };

    auto stageToLDS = [&](int buf) {
#pragma unroll
        for (int g = 0; g < GPT; ++g) {
            const int G    = g * THREADS + tid;
            const int row  = G >> 3;
            const int kofs = (G & 7) * 8;
            v8h pa, pb;
#pragma unroll
            for (int e = 0; e < 8; ++e) {
                pa[e] = (_Float16)areg[g][e];
                pb[e] = (_Float16)(float)breg[g][e];   // exact: |w| <= 127
            }
            *(v8h*)&As[buf][row * LDT + kofs] = pa;   // ds_store_b128
            *(v8h*)&Bs[buf][row * LDT + kofs] = pb;
        }
    };

    v8f acc[2][4] = {};

    loadTiles(0);
    stageToLDS(0);
    int buf = 0;

    for (int kt = 0; kt < KT; ++kt) {
        __syncthreads();                 // buf's tile visible; buf^1's readers done

        if (kt + 1 < KT) loadTiles(kt + 1);          // globals overlap WMMAs below
        if (kt + 2 < KT) {                           // warm L2 two tiles ahead
            const int prow = tid >> 1, pk = (kt + 2) * BK + (tid & 1) * 32;
            __builtin_prefetch(xg + (size_t)(mBase + prow) * Ktot + pk, 0, 0);
            __builtin_prefetch(wg + (size_t)(nBase + prow) * Ktot + pk, 0, 0);
        }

        // per-lane fragment base pointers (both A and B are K-contiguous per lane)
        const _Float16* Aw = &As[buf][(waveM * 32 + lrow) * LDT + half * 8];
        const _Float16* Bw = &Bs[buf][(waveN * 64 + lrow) * LDT + half * 8];

#pragma unroll
        for (int kk = 0; kk < BK; kk += 32) {
            v16h af[2], bf[4];
#pragma unroll
            for (int i = 0; i < 2; ++i) {
                const _Float16* p = Aw + i * 16 * LDT + kk;
                v8h lo = *(const v8h*)p;
                v8h hi = *(const v8h*)(p + 16);
                af[i] = __builtin_shufflevector(lo, hi,
                        0,1,2,3,4,5,6,7,8,9,10,11,12,13,14,15);
            }
#pragma unroll
            for (int j = 0; j < 4; ++j) {
                const _Float16* p = Bw + j * 16 * LDT + kk;
                v8h lo = *(const v8h*)p;
                v8h hi = *(const v8h*)(p + 16);
                bf[j] = __builtin_shufflevector(lo, hi,
                        0,1,2,3,4,5,6,7,8,9,10,11,12,13,14,15);
            }
#pragma unroll
            for (int i = 0; i < 2; ++i)
#pragma unroll
                for (int j = 0; j < 4; ++j)
                    acc[i][j] = __builtin_amdgcn_wmma_f32_16x16x32_f16(
                        /*neg_a=*/false, af[i], /*neg_b=*/false, bf[j],
                        /*c_mod=*/(short)0, acc[i][j],
                        /*reuse_a=*/false, /*reuse_b=*/false);
        }

        // stage tile kt+1 into the idle buffer; its previous readers finished
        // before this iteration's barrier, so no second barrier is needed.
        if (kt + 1 < KT) stageToLDS(buf ^ 1);
        buf ^= 1;
    }

    // epilogue: y = acc*scale[col] + bias[col]
    // C/D layout: VGPR v -> row = v + (lane>=16 ? 8 : 0); col = lane%16
#pragma unroll
    for (int j = 0; j < 4; ++j) {
        const int col = nBase + waveN * 64 + j * 16 + lrow;
        const float sc = scale[col];
        const float bi = bias[col];
#pragma unroll
        for (int i = 0; i < 2; ++i) {
            const int rbase = mBase + waveM * 32 + i * 16 + half * 8;
            v8f c = acc[i][j];
#pragma unroll
            for (int v = 0; v < 8; ++v)
                out[(size_t)(rbase + v) * Ntot + col] = c[v] * sc + bi;
        }
    }
}

extern "C" void kernel_launch(void* const* d_in, const int* in_sizes, int n_in,
                              void* d_out, int out_size, void* d_ws, size_t ws_size,
                              hipStream_t stream) {
    const float* x     = (const float*)d_in[0];
    const int*   w8    = (const int*)  d_in[1];
    const float* scale = (const float*)d_in[2];
    const float* bias  = (const float*)d_in[3];
    float*       out   = (float*)d_out;

    dim3 grid(Ntot / BN, Mtot / BM);   // 32 x 64 = 2048 blocks
    qlinear_wmma_f16<<<grid, THREADS, 0, stream>>>(x, w8, scale, bias, out);
}
